// SelfAttention_45896020525605
// MI455X (gfx1250) — matrix operand order repaired and checked by reading, not compiled, not run on hardware
//
#include <hip/hip_runtime.h>

// ---------------------------------------------------------------------------
// Self-attention for MI455X (gfx1250, wave32, WMMA, async-to-LDS).
// out = gamma * softmax(QK^T/sqrt(Hd)) V + x,  Q=xWq+bq, K=xWk+bk, V=xWv+bv
// B=8, S=2048, D=1024, Hd=256. All f16 WMMA with f32 accumulation.
// ---------------------------------------------------------------------------

typedef _Float16 half_t;
typedef __attribute__((ext_vector_type(16))) _Float16 v16h;
typedef __attribute__((ext_vector_type(8)))  _Float16 v8h;
typedef __attribute__((ext_vector_type(4)))  _Float16 v4h;
typedef __attribute__((ext_vector_type(8)))  float    v8f;

#define B_DIM   8
#define S_LEN   2048
#define D_DIM   1024
#define HD      256
#define M_TOT   (B_DIM * S_LEN)   // 16384

union Frag16 { v16h v; v8h h[2]; };

__device__ __forceinline__ v8f zero8() {
  v8f z = {0.f, 0.f, 0.f, 0.f, 0.f, 0.f, 0.f, 0.f};
  return z;
}

__device__ __forceinline__ v8f wmma_f16(v16h a, v16h b, v8f c) {
  return __builtin_amdgcn_wmma_f32_16x16x32_f16(false, a, false, b, (short)0, c,
                                                false, false);
}

// Scheduling fence: keep fragment-load clusters above the WMMA chain so the
// allocator cannot re-serialize them into one register pair.
__device__ __forceinline__ void sched_fence() { __builtin_amdgcn_sched_barrier(0); }

// A fragment (16x32 f16, MxK): halfs 0..7 -> K = g*8+[0,8); 8..15 -> 16+g*8+[0,8)
__device__ __forceinline__ v16h load_a_frag(const half_t* p, int g) {
  Frag16 f;
  f.h[0] = *(const v8h*)(p + g * 8);
  f.h[1] = *(const v8h*)(p + 16 + g * 8);
  return f.v;
}

// B fragment (32x16 f16, KxN): halfs 0..7 -> K = g*8+[0,8); 8..15 -> 16+g*8+[0,8)
__device__ __forceinline__ v16h load_b_frag(const half_t* p, int g) {
  Frag16 f;
  f.h[0] = *(const v8h*)(p + g * 8);
  f.h[1] = *(const v8h*)(p + 16 + g * 8);
  return f.v;
}

// gfx1250 async global->LDS copy (16B per lane), tracked by ASYNCcnt.
__device__ __forceinline__ void async_load_b128(unsigned lds_addr, unsigned gofs,
                                                const void* sbase) {
  asm volatile("global_load_async_to_lds_b128 %0, %1, %2"
               :: "v"(lds_addr), "v"(gofs), "s"(sbase)
               : "memory");
}

__device__ __forceinline__ void wait_async0() {
  asm volatile("s_wait_asynccnt 0x0" ::: "memory");
}

// ---------------------------------------------------------------------------
// f32 -> f16 conversion, 4 elems/thread
// ---------------------------------------------------------------------------
__global__ __launch_bounds__(256) void cvt_f32_f16(const float* __restrict__ in,
                                                   half_t* __restrict__ out,
                                                   int n4) {
  int i = blockIdx.x * blockDim.x + threadIdx.x;
  if (i < n4) {
    float4 f = ((const float4*)in)[i];
    v4h h = { (_Float16)f.x, (_Float16)f.y, (_Float16)f.z, (_Float16)f.w };
    ((v4h*)out)[i] = h;
  }
}

// f32 [K][N] -> f16 transposed [N][K] (one-time, weights only)
__global__ __launch_bounds__(256)
void cvt_transpose_f16(const float* __restrict__ in, half_t* __restrict__ out,
                       int K, int N) {
  int idx = blockIdx.x * blockDim.x + threadIdx.x;
  if (idx < N * K) {
    int n = idx / K;
    int k = idx - n * K;          // consecutive tid -> consecutive k: coalesced writes
    out[idx] = (half_t)in[(size_t)k * N + n];
  }
}

// ---------------------------------------------------------------------------
// GEMM: C[M,N] = A[M,K] * W^T[N,K]^T + bias, f16 in/out, f32 accum.
// Block 128x128, 8 waves as 4(M)x2(N), each wave 32x64 (2x4 WMMA tiles).
// Double-buffered LDS fed by global_load_async_to_lds_b128.
// TOUT=1: write V transposed per batch: C[b*D*S + n*S + (m & (S-1))]
// ---------------------------------------------------------------------------
#define GT_M 128
#define GT_N 128
#define GT_K 32
#define LDSW 40                       // padded stride (halfs) -> 80B rows
#define TILE_HALFS (GT_M * LDSW)      // 5120 halfs = 10240 B per buffer

template <int TOUT>
__global__ __launch_bounds__(256)
void gemm_proj(const half_t* __restrict__ A,   // [M, K]
               const half_t* __restrict__ Wt,  // [N, K] (pre-transposed)
               const float* __restrict__ bias, // [N]
               half_t* __restrict__ C,
               int N, int K) {
  __shared__ __align__(16) half_t ldsA[2][TILE_HALFS];
  __shared__ __align__(16) half_t ldsB[2][TILE_HALFS];

  const int tid  = threadIdx.x;
  const int wave = tid >> 5;
  const int lane = tid & 31;
  const int g    = lane >> 4;
  const int ln   = lane & 15;
  const int mBlk = blockIdx.y * GT_M;
  const int nBlk = blockIdx.x * GT_N;
  const int wm   = wave >> 1;
  const int wn   = wave & 1;

  const unsigned aBase = (unsigned)(uintptr_t)&ldsA[0][0];
  const unsigned bBase = (unsigned)(uintptr_t)&ldsB[0][0];

  // Stage one 128x32 K-slice of A and Wt into LDS buffer `buf` (async).
  auto stage = [&](int buf, int k0) {
#pragma unroll
    for (int i = 0; i < 2; ++i) {
      const int c   = tid + i * 256;    // 512 chunks of 16B per tile
      const int row = c >> 2;
      const int cc  = c & 3;
      const unsigned lOff = (unsigned)(buf * TILE_HALFS + row * LDSW) * 2u + cc * 16u;
      async_load_b128(aBase + lOff,
                      ((unsigned)(mBlk + row) * K + k0 + cc * 8) * 2u, A);
      async_load_b128(bBase + lOff,
                      ((unsigned)(nBlk + row) * K + k0 + cc * 8) * 2u, Wt);
    }
  };

  v8f acc[2][4];
#pragma unroll
  for (int i = 0; i < 2; ++i)
#pragma unroll
    for (int j = 0; j < 4; ++j) acc[i][j] = zero8();

  const int KT = K / GT_K;
  stage(0, 0);

#pragma unroll 1
  for (int kt = 0; kt < KT; ++kt) {
    const int cur = kt & 1;
    wait_async0();       // my async copies for buffer `cur` are done
    __syncthreads();     // everyone's copies done; everyone done reading cur^1
    if (kt + 1 < KT) stage(cur ^ 1, (kt + 1) * GT_K);

#pragma unroll
    for (int i = 0; i < 2; ++i) {
      v16h a = load_a_frag(&ldsA[cur][(wm * 32 + i * 16 + ln) * LDSW], g);
#pragma unroll
      for (int j = 0; j < 4; ++j) {
        v16h b = load_b_frag(&ldsB[cur][(wn * 64 + j * 16 + ln) * LDSW], g);
        acc[i][j] = wmma_f16(a, b, acc[i][j]);
      }
    }
  }

  // Epilogue: C element (M = e + 8g, N = ln)
#pragma unroll
  for (int i = 0; i < 2; ++i) {
#pragma unroll
    for (int j = 0; j < 4; ++j) {
      const int n = nBlk + wn * 64 + j * 16 + ln;
      const float bv = bias[n];
#pragma unroll
      for (int e = 0; e < 8; ++e) {
        const int m = mBlk + wm * 32 + i * 16 + 8 * g + e;
        const float val = acc[i][j][e] + bv;
        if (TOUT == 0) {
          C[(size_t)m * N + n] = (half_t)val;
        } else {  // V^T per batch: [b][D][S]
          const int b = m >> 11;
          const int s = m & (S_LEN - 1);
          C[((size_t)b * N + n) * S_LEN + s] = (half_t)val;
        }
      }
    }
  }
}

// ---------------------------------------------------------------------------
// Pass 1: per-row softmax stats (max, 1/sum) via WMMA QK^T.
// grid = B*S/64 blocks of 4 waves; each wave owns 16 query rows.
// 32 keys per iteration: both fragment sets (32 global_load_b128) issue ahead
// of the WMMAs (sched_barrier); c0's chain overlaps bf1's load latency.
// ---------------------------------------------------------------------------
__global__ __launch_bounds__(128)
void softmax_stats(const half_t* __restrict__ Qh, const half_t* __restrict__ Kh,
                   float* __restrict__ Mrow, float* __restrict__ Sinv) {
  const int tid  = threadIdx.x;
  const int wave = tid >> 5;
  const int lane = tid & 31;
  const int g    = lane >> 4;
  const int ln   = lane & 15;
  const int blk  = blockIdx.x;
  const int b    = blk >> 5;                 // 32 blocks of 64 rows per batch
  const int qB   = (blk & 31) * 64 + wave * 16;

  const half_t* qptr = Qh + ((size_t)b * S_LEN + qB) * HD;
  const half_t* kptr = Kh + (size_t)b * S_LEN * HD;

  v16h qa[8];
#pragma unroll
  for (int t = 0; t < 8; ++t) qa[t] = load_a_frag(qptr + (size_t)ln * HD + t * 32, g);

  float m[8], s[8];
#pragma unroll
  for (int e = 0; e < 8; ++e) { m[e] = -1e30f; s[e] = 0.f; }

  const float scale = 0.0625f;  // Hd^-0.5

#pragma unroll 1
  for (int kv = 0; kv < S_LEN; kv += 32) {
    const half_t* krow0 = kptr + (size_t)(kv + ln) * HD;
    const half_t* krow1 = krow0 + (size_t)16 * HD;
    v16h bf0[8], bf1[8];
#pragma unroll
    for (int t = 0; t < 8; ++t) bf0[t] = load_b_frag(krow0 + t * 32, g);
#pragma unroll
    for (int t = 0; t < 8; ++t) bf1[t] = load_b_frag(krow1 + t * 32, g);
    sched_fence();
    v8f c0 = zero8();
    v8f c1 = zero8();
#pragma unroll
    for (int t = 0; t < 8; ++t) c0 = wmma_f16(qa[t], bf0[t], c0);
#pragma unroll
    for (int t = 0; t < 8; ++t) c1 = wmma_f16(qa[t], bf1[t], c1);
#pragma unroll
    for (int e = 0; e < 8; ++e) {
      float e0 = c0[e] * scale;
      float e1 = c1[e] * scale;
      float mn = fmaxf(m[e], fmaxf(e0, e1));
      s[e] = s[e] * __expf(m[e] - mn) + __expf(e0 - mn) + __expf(e1 - mn);
      m[e] = mn;
    }
  }

  // merge (m,s) across the 16 lanes of each half-wave
#pragma unroll
  for (int e = 0; e < 8; ++e) {
    float mv = m[e], sv = s[e];
#pragma unroll
    for (int mask = 1; mask < 16; mask <<= 1) {
      float mo = __shfl_xor(mv, mask, 32);
      float so = __shfl_xor(sv, mask, 32);
      float mn = fmaxf(mv, mo);
      sv = sv * __expf(mv - mn) + so * __expf(mo - mn);
      mv = mn;
    }
    m[e] = mv; s[e] = sv;
  }

  if (ln == 0) {
#pragma unroll
    for (int e = 0; e < 8; ++e) {
      const int row = b * S_LEN + qB + 8 * g + e;
      Mrow[row] = m[e];
      Sinv[row] = 1.0f / s[e];
    }
  }
}

// ---------------------------------------------------------------------------
// Pass 2: out = gamma * (P V) + x. grid = B*S/16, 8 waves/block.
// ---------------------------------------------------------------------------
#define PSTR 136  // padded LDS stride for P (halfs)

__global__ __launch_bounds__(256)
void attn_out(const half_t* __restrict__ Qh, const half_t* __restrict__ Kh,
              const half_t* __restrict__ Vt, const float* __restrict__ Mrow,
              const float* __restrict__ Sinv, const float* __restrict__ x,
              const float* __restrict__ gamma, float* __restrict__ out) {
  __shared__ __align__(16) half_t ldsP[16 * PSTR];

  const int tid  = threadIdx.x;
  const int wave = tid >> 5;
  const int lane = tid & 31;
  const int g    = lane >> 4;
  const int ln   = lane & 15;
  const int blk  = blockIdx.x;
  const int b    = blk >> 7;
  const int qB   = (blk & 127) * 16;

  const half_t* qptr = Qh + ((size_t)b * S_LEN + qB) * HD;
  const half_t* kptr = Kh + (size_t)b * S_LEN * HD;
  const half_t* vptr = Vt + (size_t)b * D_DIM * S_LEN;  // [D][S]

  v16h qa[8];
#pragma unroll
  for (int t = 0; t < 8; ++t) qa[t] = load_a_frag(qptr + (size_t)ln * HD + t * 32, g);

  float mr[8], is[8];
#pragma unroll
  for (int e = 0; e < 8; ++e) {
    const int row = b * S_LEN + qB + 8 * g + e;
    mr[e] = Mrow[row];
    is[e] = Sinv[row];
  }

  v8f acc[8];
#pragma unroll
  for (int j = 0; j < 8; ++j) acc[j] = zero8();

  const float scale = 0.0625f;
  const float gm = gamma[0];

#pragma unroll 1
  for (int kv = 0; kv < S_LEN; kv += 128) {
    if (kv + 128 < S_LEN) {
      __builtin_prefetch(kptr + (size_t)(kv + 128 + wave * 16 + ln) * HD, 0, 0);
      __builtin_prefetch(vptr + (size_t)(wave * 128 + ln) * S_LEN + kv + 128, 0, 0);
    }
    // Phase 1: this wave's 16x16 E tile at keys kv + wave*16
    {
      const half_t* krow = kptr + (size_t)(kv + wave * 16 + ln) * HD;
      v16h bf[8];
#pragma unroll
      for (int t = 0; t < 8; ++t) bf[t] = load_b_frag(krow + t * 32, g);
      sched_fence();
      v8f c = zero8();
#pragma unroll
      for (int t = 0; t < 8; ++t) c = wmma_f16(qa[t], bf[t], c);
#pragma unroll
      for (int e = 0; e < 8; ++e) {
        float p = __expf(c[e] * scale - mr[e]) * is[e];
        ldsP[(8 * g + e) * PSTR + wave * 16 + ln] = (half_t)p;
      }
    }
    __syncthreads();

    // Phase 2: acc += P(16x128) * V(128 x this wave's 128 d-cols).
#pragma unroll
    for (int t = 0; t < 4; ++t) {
      v16h pa = load_a_frag(&ldsP[ln * PSTR + t * 32], g);
#pragma unroll
      for (int jh = 0; jh < 2; ++jh) {
        v16h vb[4];
#pragma unroll
        for (int u = 0; u < 4; ++u) {
          const int dcol = wave * 128 + (jh * 4 + u) * 16 + ln;
          vb[u] = load_b_frag(vptr + (size_t)dcol * S_LEN + kv + t * 32, g);
        }
        sched_fence();
#pragma unroll
        for (int u = 0; u < 4; ++u)
          acc[jh * 4 + u] = wmma_f16(pa, vb[u], acc[jh * 4 + u]);
      }
    }
    __syncthreads();
  }

  // Epilogue: out = gamma*acc + x
#pragma unroll
  for (int j = 0; j < 8; ++j) {
#pragma unroll
    for (int e = 0; e < 8; ++e) {
      const size_t row = (size_t)b * S_LEN + qB + 8 * g + e;
      const size_t idx = row * D_DIM + wave * 128 + j * 16 + ln;
      out[idx] = gm * acc[j][e] + x[idx];
    }
  }
}

// ---------------------------------------------------------------------------
// Launch
// ---------------------------------------------------------------------------
extern "C" void kernel_launch(void* const* d_in, const int* in_sizes, int n_in,
                              void* d_out, int out_size, void* d_ws, size_t ws_size,
                              hipStream_t stream) {
  const float* x     = (const float*)d_in[0];
  const float* Wq    = (const float*)d_in[1];
  const float* bq    = (const float*)d_in[2];
  const float* Wk    = (const float*)d_in[3];
  const float* bk    = (const float*)d_in[4];
  const float* Wv    = (const float*)d_in[5];
  const float* bv    = (const float*)d_in[6];
  const float* gamma = (const float*)d_in[7];
  float* out = (float*)d_out;

  size_t off = 0;
  auto take = [&](size_t bytes) -> void* {
    void* p = (char*)d_ws + off;
    off += (bytes + 255) & ~(size_t)255;
    return p;
  };

  half_t* xh   = (half_t*)take((size_t)M_TOT * D_DIM * 2);   // 32 MB
  half_t* Wqt  = (half_t*)take((size_t)D_DIM * HD * 2);      // [N][K] f16
  half_t* Wkt  = (half_t*)take((size_t)D_DIM * HD * 2);
  half_t* Wvt  = (half_t*)take((size_t)D_DIM * D_DIM * 2);
  half_t* Qh   = (half_t*)take((size_t)M_TOT * HD * 2);      // 8 MB
  half_t* Kh   = (half_t*)take((size_t)M_TOT * HD * 2);      // 8 MB
  half_t* Vth  = (half_t*)take((size_t)M_TOT * D_DIM * 2);   // 32 MB (transposed)
  float*  Mrow = (float*)take((size_t)M_TOT * 4);
  float*  Sinv = (float*)take((size_t)M_TOT * 4);

  // 1) conversions: x -> f16; weights -> f16 transposed [N][K]
  {
    int n4 = (M_TOT * D_DIM) / 4;
    cvt_f32_f16<<<(n4 + 255) / 256, 256, 0, stream>>>(x, xh, n4);
    int nq = D_DIM * HD;
    cvt_transpose_f16<<<(nq + 255) / 256, 256, 0, stream>>>(Wq, Wqt, D_DIM, HD);
    cvt_transpose_f16<<<(nq + 255) / 256, 256, 0, stream>>>(Wk, Wkt, D_DIM, HD);
    int nv = D_DIM * D_DIM;
    cvt_transpose_f16<<<(nv + 255) / 256, 256, 0, stream>>>(Wv, Wvt, D_DIM, D_DIM);
  }

  // 2) QKV projections (V written transposed per batch)
  gemm_proj<0><<<dim3(HD / GT_N, M_TOT / GT_M),    256, 0, stream>>>(xh, Wqt, bq, Qh,  HD,    D_DIM);
  gemm_proj<0><<<dim3(HD / GT_N, M_TOT / GT_M),    256, 0, stream>>>(xh, Wkt, bk, Kh,  HD,    D_DIM);
  gemm_proj<1><<<dim3(D_DIM / GT_N, M_TOT / GT_M), 256, 0, stream>>>(xh, Wvt, bv, Vth, D_DIM, D_DIM);

  // 3) softmax row stats
  softmax_stats<<<B_DIM * (S_LEN / 64), 128, 0, stream>>>(Qh, Kh, Mrow, Sinv);

  // 4) attention output, fused gamma*out + x
  attn_out<<<B_DIM * (S_LEN / 16), 256, 0, stream>>>(Qh, Kh, Vth, Mrow, Sinv, x, gamma, out);
}
